// NeighborhoodAttention2D_16406775071495
// MI455X (gfx1250) — matrix operand order repaired and checked by reading, not compiled, not run on hardware
//
#include <hip/hip_runtime.h>

typedef __attribute__((ext_vector_type(16))) __bf16  v16bf;
typedef __attribute__((ext_vector_type(8)))  float   v8f;

#define DIM     256
#define NHEADS  8
#define HDIM    32
#define BATCH   4
#define HH      56
#define WW      56
#define NN      (HH * WW)        // 3136
#define KSIZE   7
#define WCLIP   (WW - KSIZE)     // 49

__device__ __forceinline__ v8f wmma_bf16(v16bf a, v16bf b, v8f c) {
    // D = A(16x32 bf16) * B(32x16 bf16) + C(16x16 f32)
    return __builtin_amdgcn_wmma_f32_16x16x32_bf16(false, a, false, b, (short)0, c,
                                                   false, false);
}

// ---------------------------------------------------------------------------
// Kernel 0: weight prep. Transpose+convert both weight matrices to bf16,
// K-major (Wt[col][k]) so GEMM B fragments are contiguous 2x b128 loads.
// ---------------------------------------------------------------------------
__global__ __launch_bounds__(256)
void wprep_kernel(const float* __restrict__ wqkv, const float* __restrict__ wproj,
                  __bf16* __restrict__ wqkvT, __bf16* __restrict__ wprojT) {
    const int i = blockIdx.x * 256 + threadIdx.x;
    if (i < 768 * 256) {
        const int col = i >> 8;          // 0..767
        const int k   = i & 255;
        wqkvT[i] = (__bf16)wqkv[(size_t)k * 768 + col];
    } else {
        const int j = i - 768 * 256;     // 0..65535
        const int col = j >> 8;
        const int k   = j & 255;
        wprojT[j] = (__bf16)wproj[(size_t)k * 256 + col];
    }
}

// ---------------------------------------------------------------------------
// Kernel 1: QKV projection GEMM.  X[12544,256] f32 x WqT[768,256] bf16
// -> q (scaled), k, v in bf16 workspace laid out [b, head, n, d].
// One wave per 16x64 output strip: A fragment reused across 4 WMMAs.
// ---------------------------------------------------------------------------
__global__ __launch_bounds__(256)
void qkv_gemm_kernel(const float* __restrict__ X, const __bf16* __restrict__ WqT,
                     const float* __restrict__ Bq,
                     __bf16* __restrict__ qws, __bf16* __restrict__ kws,
                     __bf16* __restrict__ vws) {
    const int lane = threadIdx.x & 31;
    const int wave = threadIdx.x >> 5;
    const int tile = blockIdx.x * 8 + wave;      // 784 * 12 strips
    const int tm = tile / 12;
    const int tn = tile - tm * 12;
    const int m0 = tm * 16, n0 = tn * 64;
    const int ll = lane & 15, lh = lane >> 4;

    v8f acc[4] = {{}, {}, {}, {}};
    const float*  xrow  = X + (size_t)(m0 + ll) * DIM;
    const __bf16* wbase = WqT + (size_t)(n0 + ll) * DIM;
    #pragma unroll
    for (int kk = 0; kk < 8; ++kk) {
        // A layout: M = lane&15; lanes0-15 K=0..15, lanes16-31 K=16..31
        v16bf a;
        const float* xp = xrow + kk * 32 + lh * 16;
        // WGP-scope prefetch of next K-step's A line (data is L2-resident;
        // locality 3 -> near-scope prefetch pulls into WGP cache).
        if (kk < 7) __builtin_prefetch(xp + 32, 0, 3);
        #pragma unroll
        for (int j = 0; j < 8; ++j) { a[j] = (__bf16)xp[j]; a[8 + j] = (__bf16)xp[8 + j]; }
        #pragma unroll
        for (int t = 0; t < 4; ++t) {
            // B layout: N = lane&15; lanes0-15 K=0..15, lanes16-31 K=16..31 (contiguous!)
            v16bf b;
            const __bf16* wp = wbase + (size_t)t * 16 * DIM + kk * 32 + lh * 16;
            #pragma unroll
            for (int j = 0; j < 16; ++j) b[j] = wp[j];
            acc[t] = wmma_bf16(a, b, acc[t]);
        }
    }

    #pragma unroll
    for (int t = 0; t < 4; ++t) {
        const int gcol  = n0 + t * 16 + ll;      // 0..767
        const float bias = Bq[gcol];
        const int which = gcol >> 8;             // 0=q 1=k 2=v
        const int rem   = gcol & 255;
        const int head  = rem >> 5;
        const int dd    = rem & 31;
        #pragma unroll
        for (int r = 0; r < 8; ++r) {            // C layout: M = r + 8*lh, N = lane&15
            const int grow = m0 + r + lh * 8;
            const int bb = grow / NN;
            const int nn = grow - bb * NN;
            const size_t idx = ((size_t)(bb * NHEADS + head) * NN + nn) * HDIM + dd;
            const float val = acc[t][r] + bias;
            if (which == 0)      qws[idx] = (__bf16)(val * 0.17677669529663689f); // 1/sqrt(32)
            else if (which == 1) kws[idx] = (__bf16)val;
            else                 vws[idx] = (__bf16)val;
        }
    }
}

// ---------------------------------------------------------------------------
// Kernel 2: neighborhood attention, flash-style over the union key window.
// One wave handles 16 queries of one image row (one (b,head,row,coltile)).
// Union window: 7 rows x <=22 cols <= 154 keys -> 5 chunks of 32 keys.
// Per chunk: 2x QK^T WMMA, mask, online softmax, 2x PV WMMA.
// ---------------------------------------------------------------------------
__global__ __launch_bounds__(256)
void natten_kernel(const __bf16* __restrict__ qws, const __bf16* __restrict__ kws,
                   const __bf16* __restrict__ vws, __bf16* __restrict__ aws) {
    __shared__ __bf16 ldsP[8][16][32];   // wave-private P tiles (C-layout -> A-layout bounce)

    const int lane = threadIdx.x & 31;
    const int wave = threadIdx.x >> 5;
    const int tid  = blockIdx.x * 8 + wave;
    const int ct   = tid & 3;            // column tile (16 queries)
    const int t1   = tid >> 2;
    const int qrow = t1 % HH;
    const int bh   = t1 / HH;            // b*8 + head
    const int ll = lane & 15, lh = lane >> 4;

    const int c0     = ct * 16;
    const int cq_hi  = min(c0 + 15, WW - 1);
    const int col_lo = min(max(c0 - 3, 0), WCLIP);
    const int col_hi = min(max(cq_hi - 3, 0), WCLIP) + (KSIZE - 1);
    const int ncols  = col_hi - col_lo + 1;          // <= 22
    const int row_lo = min(max(qrow - 3, 0), WCLIP); // whole row window valid for tile
    const int nkeys  = KSIZE * ncols;                // <= 154

    const __bf16* qb = qws + (size_t)bh * NN * HDIM;
    const __bf16* kb = kws + (size_t)bh * NN * HDIM;
    const __bf16* vb = vws + (size_t)bh * NN * HDIM;

    // Q tile in WMMA A layout (16 queries x 32 channels)
    v16bf aq = {};
    {
        const int qc = c0 + ll;
        if (qc < WW) {
            const __bf16* qp = qb + ((size_t)(qrow * WW + qc)) * HDIM + lh * 16;
            #pragma unroll
            for (int j = 0; j < 8; ++j) { aq[j] = qp[j]; aq[8 + j] = qp[8 + j]; }
        }
    }

    float mrow[8], lrow[8];
    #pragma unroll
    for (int r = 0; r < 8; ++r) { mrow[r] = -1e30f; lrow[r] = 0.f; }
    v8f olo = {}, ohi = {};

    for (int chk = 0; chk < 5; ++chk) {
        const int kb0 = chk * 32;

        // ---- K tiles in B layout (two 16-key tiles) ----
        v16bf bk0 = {}, bk1 = {};
        int kc0, kc1; bool kv0, kv1;
        {
            const int ki = kb0 + ll;
            kv0 = ki < nkeys;
            const int wr = ki / ncols, wc = ki - wr * ncols;
            kc0 = col_lo + wc;
            if (kv0) {
                const __bf16* kp = kb + ((size_t)((row_lo + wr) * WW + kc0)) * HDIM + lh * 16;
                #pragma unroll
                for (int j = 0; j < 16; ++j) bk0[j] = kp[j];
            }
        }
        {
            const int ki = kb0 + 16 + ll;
            kv1 = ki < nkeys;
            const int wr = ki / ncols, wc = ki - wr * ncols;
            kc1 = col_lo + wc;
            if (kv1) {
                const __bf16* kp = kb + ((size_t)((row_lo + wr) * WW + kc1)) * HDIM + lh * 16;
                #pragma unroll
                for (int j = 0; j < 16; ++j) bk1[j] = kp[j];
            }
            // WGP-scope prefetch of next chunk's key row for this lane
            const int kin = ki + 16;
            if (kin < nkeys) {
                const int wrn = kin / ncols, wcn = kin - wrn * ncols;
                __builtin_prefetch(kb + ((size_t)((row_lo + wrn) * WW + col_lo + wcn)) * HDIM,
                                   0, 3);
            }
        }
        v8f z = {};
        v8f s0 = wmma_bf16(aq, bk0, z);
        v8f s1 = wmma_bf16(aq, bk1, z);

        // ---- mask + online softmax (row stats live in 16-lane halves) ----
        #pragma unroll
        for (int r = 0; r < 8; ++r) {
            const int qc_r = c0 + r + lh * 8;                 // query of this C row
            const int qs   = min(max(qc_r - 3, 0), WCLIP);    // query's window start col
            const bool qv  = qc_r < WW;
            float x0 = (qv && kv0 && kc0 >= qs && kc0 <= qs + 6) ? s0[r] : -1e30f;
            float x1 = (qv && kv1 && kc1 >= qs && kc1 <= qs + 6) ? s1[r] : -1e30f;
            float m = fmaxf(x0, x1);
            m = fmaxf(m, __shfl_xor(m, 1, 32));
            m = fmaxf(m, __shfl_xor(m, 2, 32));
            m = fmaxf(m, __shfl_xor(m, 4, 32));
            m = fmaxf(m, __shfl_xor(m, 8, 32));
            const float mn    = fmaxf(mrow[r], m);
            const float alpha = __expf(mrow[r] - mn);
            const float p0 = (x0 > -1e29f) ? __expf(x0 - mn) : 0.f;
            const float p1 = (x1 > -1e29f) ? __expf(x1 - mn) : 0.f;
            float s = p0 + p1;
            s += __shfl_xor(s, 1, 32);
            s += __shfl_xor(s, 2, 32);
            s += __shfl_xor(s, 4, 32);
            s += __shfl_xor(s, 8, 32);
            lrow[r] = lrow[r] * alpha + s;
            mrow[r] = mn;
            olo[r] *= alpha;
            ohi[r] *= alpha;
            const int qm = r + lh * 8;
            ldsP[wave][qm][ll]      = (__bf16)p0;
            ldsP[wave][qm][16 + ll] = (__bf16)p1;
        }

        // ---- reload P in A layout (16 queries x 32 chunk-keys) ----
        v16bf pa;
        #pragma unroll
        for (int j = 0; j < 8; ++j) {
            pa[j]     = ldsP[wave][ll][lh * 16 + j];
            pa[8 + j] = ldsP[wave][ll][lh * 16 + 8 + j];
        }

        // ---- V tiles in B layout (32 chunk-keys x 16 channels, two halves) ----
        v16bf bv0 = {}, bv1 = {};
        {
            int ki = kb0 + lh * 16;
            int wr = ki / ncols, wc = ki - wr * ncols;
            #pragma unroll
            for (int j = 0; j < 16; ++j) {
                if (ki < nkeys) {
                    const size_t off = ((size_t)((row_lo + wr) * WW + col_lo + wc)) * HDIM;
                    bv0[j] = vb[off + ll];
                    bv1[j] = vb[off + 16 + ll];
                }
                ++ki; ++wc; if (wc == ncols) { wc = 0; ++wr; }
            }
        }
        olo = wmma_bf16(pa, bv0, olo);
        ohi = wmma_bf16(pa, bv1, ohi);
    }

    // ---- normalize + scatter to [b, n, head*32+ch] bf16 for the proj GEMM ----
    #pragma unroll
    for (int r = 0; r < 8; ++r) {
        const int qc_r = c0 + r + lh * 8;
        if (qc_r < WW) {
            const float inv = 1.0f / lrow[r];
            const int bb = bh >> 3, head = bh & 7;
            const size_t base = ((size_t)bb * NN + (qrow * WW + qc_r)) * DIM + head * HDIM;
            aws[base + ll]      = (__bf16)(olo[r] * inv);
            aws[base + 16 + ll] = (__bf16)(ohi[r] * inv);
        }
    }
}

// ---------------------------------------------------------------------------
// Kernel 3: output projection.  A[12544,256] bf16 x WpT[256,256] bf16 -> f32.
// One wave per 16x64 output strip.
// ---------------------------------------------------------------------------
__global__ __launch_bounds__(256)
void proj_gemm_kernel(const __bf16* __restrict__ A, const __bf16* __restrict__ WpT,
                      const float* __restrict__ Bp, float* __restrict__ Out) {
    const int lane = threadIdx.x & 31;
    const int wave = threadIdx.x >> 5;
    const int tile = blockIdx.x * 8 + wave;      // 784 * 4 strips
    const int tm = tile >> 2;
    const int tn = tile & 3;
    const int m0 = tm * 16, n0 = tn * 64;
    const int ll = lane & 15, lh = lane >> 4;

    v8f acc[4] = {{}, {}, {}, {}};
    const __bf16* arow  = A + (size_t)(m0 + ll) * DIM;
    const __bf16* wbase = WpT + (size_t)(n0 + ll) * DIM;
    #pragma unroll
    for (int kk = 0; kk < 8; ++kk) {
        v16bf a;
        const __bf16* ap = arow + kk * 32 + lh * 16;
        if (kk < 7) __builtin_prefetch(ap + 32, 0, 3);  // WGP-scope prefetch, next K-step
        #pragma unroll
        for (int j = 0; j < 8; ++j) { a[j] = ap[j]; a[8 + j] = ap[8 + j]; }
        #pragma unroll
        for (int t = 0; t < 4; ++t) {
            v16bf b;
            const __bf16* wp = wbase + (size_t)t * 16 * DIM + kk * 32 + lh * 16;
            #pragma unroll
            for (int j = 0; j < 16; ++j) b[j] = wp[j];
            acc[t] = wmma_bf16(a, b, acc[t]);
        }
    }
    #pragma unroll
    for (int t = 0; t < 4; ++t) {
        const float bias = Bp[n0 + t * 16 + ll];
        #pragma unroll
        for (int r = 0; r < 8; ++r) {
            const int grow = m0 + r + lh * 8;
            Out[(size_t)grow * DIM + n0 + t * 16 + ll] = acc[t][r] + bias;
        }
    }
}

// ---------------------------------------------------------------------------
extern "C" void kernel_launch(void* const* d_in, const int* in_sizes, int n_in,
                              void* d_out, int out_size, void* d_ws, size_t ws_size,
                              hipStream_t stream) {
    const float* x     = (const float*)d_in[0];
    const float* wqkv  = (const float*)d_in[1];
    const float* bqkv  = (const float*)d_in[2];
    const float* wproj = (const float*)d_in[3];
    const float* bproj = (const float*)d_in[4];
    float* out = (float*)d_out;

    // workspace: q | k | v | attn_out (bf16, 6.4MB each) | wqkvT | wprojT
    const size_t QKV_BYTES = (size_t)BATCH * NHEADS * NN * HDIM * 2; // 6,422,528 B
    char* ws = (char*)d_ws;
    __bf16* qws    = (__bf16*)(ws);
    __bf16* kws    = (__bf16*)(ws + QKV_BYTES);
    __bf16* vws    = (__bf16*)(ws + 2 * QKV_BYTES);
    __bf16* aws    = (__bf16*)(ws + 3 * QKV_BYTES);
    __bf16* wqkvT  = (__bf16*)(ws + 4 * QKV_BYTES);
    __bf16* wprojT = (__bf16*)(ws + 4 * QKV_BYTES + (size_t)768 * 256 * 2);

    // weight transpose+convert: (768*256 + 256*256) elements
    wprep_kernel<<<1024, 256, 0, stream>>>(wqkv, wproj, wqkvT, wprojT);
    // 784 M-tiles x 12 N-strips (64 cols), 8 wave-strips per block
    qkv_gemm_kernel<<<(784 * 12) / 8, 256, 0, stream>>>(x, wqkvT, bqkv, qws, kws, vws);
    // 4*8*56 rows x 4 column tiles
    natten_kernel<<<(BATCH * NHEADS * HH * 4) / 8, 256, 0, stream>>>(qws, kws, vws, aws);
    // 784 M-tiles x 4 N-strips
    proj_gemm_kernel<<<(784 * 4) / 8, 256, 0, stream>>>(aws, wprojT, bproj, out);
}
